// LowFreqTransformer_71313636983463
// MI455X (gfx1250) — matrix operand order repaired and checked by reading, not compiled, not run on hardware
//
#include <hip/hip_runtime.h>
#include <stdint.h>

// ---------------------------------------------------------------------------
// MI455X (gfx1250) implementation: bf16 WMMA GEMMs + flash attention.
// Wave32, V_WMMA_F32_16X16X32_BF16 everywhere the math is big.
// GEMM A-tiles staged with GLOBAL_LOAD_ASYNC_TO_LDS_B128 (ASYNCcnt) when the
// toolchain exposes the builtin; VGPR-copy fallback otherwise.
// Cross-attention has KV length 1 -> softmax == 1 -> Wcq/Wck are dead; the
// cross block reduces to a per-(layer,batch) vector broadcast + LayerNorm.
// ---------------------------------------------------------------------------

typedef __attribute__((ext_vector_type(16))) __bf16 v16bf;
typedef __attribute__((ext_vector_type(8)))  float  v8f;
typedef int v4i __attribute__((vector_size(16)));   // matches builtin pointee

#define D_MODEL 256
#define SEQ     256
#define BATCH   256
#define NHEAD   8
#define DHEAD   32

#if defined(__AMDGCN__) && __has_builtin(__builtin_amdgcn_global_load_async_to_lds_b128)
#define HAVE_ASYNC_LDS 1
#else
#define HAVE_ASYNC_LDS 0
#endif

#if HAVE_ASYNC_LDS
// ISA 10.2: flat shared addresses keep the LDS byte offset in addr[31:0],
// so integer truncation is the architected generic->LDS conversion.
// Builtin signature (from clang diagnostic): AS1 v4i* src, AS3 v4i* dst,
// imm offset, imm cpol.
#define AS_GLOBAL(p) ((__attribute__((address_space(1))) v4i*)(uintptr_t)(p))
#define AS_LDS(p)    ((__attribute__((address_space(3))) v4i*)(uint32_t)(uintptr_t)(p))
#endif

__device__ __forceinline__ void wait_async0() {
#if HAVE_ASYNC_LDS
#if __has_builtin(__builtin_amdgcn_s_wait_asynccnt)
  __builtin_amdgcn_s_wait_asynccnt(0);
#else
  asm volatile("s_wait_asynccnt 0x0" ::: "memory");
#endif
#endif
}

__device__ __forceinline__ uint16_t f2bf(float f) {
  uint32_t u = __float_as_uint(f);
  u += 0x7FFFu + ((u >> 16) & 1u);          // round-to-nearest-even
  return (uint16_t)(u >> 16);
}
__device__ __forceinline__ float bf2f(uint16_t h) {
  return __uint_as_float(((uint32_t)h) << 16);
}

union FragAB { v16bf v; uint32_t u[8]; };

// ---------------------------------------------------------------------------
// f32 -> bf16 weight conversion
// ---------------------------------------------------------------------------
__global__ void f32_to_bf16_k(const float* __restrict__ s,
                              uint16_t* __restrict__ d, int n) {
  int i = blockIdx.x * blockDim.x + threadIdx.x;
  if (i < n) d[i] = f2bf(s[i]);
}

// ---------------------------------------------------------------------------
// Tiny fp32 GEMM (cond-memory / cross vectors; M*N <= 64K, K <= 256)
// ---------------------------------------------------------------------------
__global__ void small_gemm_f32(const float* __restrict__ A,
                               const float* __restrict__ B,
                               const float* __restrict__ bias,
                               float* __restrict__ out,
                               int M, int K, int N, int relu) {
  int idx = blockIdx.x * blockDim.x + threadIdx.x;
  if (idx >= M * N) return;
  int m = idx / N, n = idx % N;
  float a = bias[n];
  for (int k = 0; k < K; ++k) a += A[m * K + k] * B[k * N + n];
  if (relu) a = fmaxf(a, 0.f);
  out[idx] = a;
}

// ---------------------------------------------------------------------------
// Input stage: shifted token -> Linear(2,256) -> LN -> ReLU -> +PE
// one wave per row (row = b*S + s), 8 dims per lane
// ---------------------------------------------------------------------------
__global__ __launch_bounds__(32) void input_stage(
    const float* __restrict__ x, const float* __restrict__ cond,
    const float* __restrict__ Win, const float* __restrict__ binp,
    const float* __restrict__ gin, const float* __restrict__ bein,
    const float* __restrict__ Ws, const float* __restrict__ bs,
    float* __restrict__ h, uint16_t* __restrict__ hb) {
  int row = blockIdx.x;
  int lane = threadIdx.x;
  int b = row >> 8, sIdx = row & 255;
  float sh0, sh1;
  if (sIdx == 0) {
    sh0 = bs[0]; sh1 = bs[1];
#pragma unroll
    for (int k = 0; k < 5; ++k) {
      float cv = cond[b * 5 + k];
      sh0 += cv * Ws[k * 2 + 0];
      sh1 += cv * Ws[k * 2 + 1];
    }
  } else {
    sh0 = x[((size_t)b * SEQ + sIdx - 1) * 2 + 0];
    sh1 = x[((size_t)b * SEQ + sIdx - 1) * 2 + 1];
  }
  float v[8], s = 0.f, s2 = 0.f;
#pragma unroll
  for (int j = 0; j < 8; ++j) {
    int d = lane * 8 + j;
    float p = sh0 * Win[d] + sh1 * Win[D_MODEL + d] + binp[d];
    v[j] = p; s += p; s2 += p * p;
  }
#pragma unroll
  for (int off = 1; off < 32; off <<= 1) {
    s  += __shfl_xor(s, off, 32);
    s2 += __shfl_xor(s2, off, 32);
  }
  float mean = s * (1.f / 256.f);
  float var  = s2 * (1.f / 256.f) - mean * mean;
  float rs   = rsqrtf(var + 1e-5f);
  size_t base = (size_t)row * D_MODEL + lane * 8;
#pragma unroll
  for (int j = 0; j < 8; ++j) {
    int d = lane * 8 + j;
    float y = (v[j] - mean) * rs * gin[d] + bein[d];
    y = fmaxf(y, 0.f);
    float freq = __expf(-(float)(d & ~1) * (9.210340372f / 256.f));
    float ang  = (float)sIdx * freq;
    y += (d & 1) ? __cosf(ang) : __sinf(ang);
    h[base + j]  = y;
    hb[base + j] = f2bf(y);
  }
}

// ---------------------------------------------------------------------------
// Residual + LayerNorm (one wave per row). srcShift=8 -> broadcast per batch.
// ---------------------------------------------------------------------------
__global__ __launch_bounds__(32) void ln_add_k(
    float* __restrict__ h, const float* __restrict__ src,
    const float* __restrict__ g, const float* __restrict__ be,
    uint16_t* __restrict__ hb, int srcShift) {
  int row = blockIdx.x;
  int lane = threadIdx.x;
  size_t base  = (size_t)row * D_MODEL + lane * 8;
  size_t sbase = (size_t)(srcShift ? (row >> srcShift) : row) * D_MODEL + lane * 8;
  float x[8], s = 0.f, s2 = 0.f;
#pragma unroll
  for (int j = 0; j < 8; ++j) {
    x[j] = h[base + j] + src[sbase + j];
    s += x[j]; s2 += x[j] * x[j];
  }
#pragma unroll
  for (int off = 1; off < 32; off <<= 1) {
    s  += __shfl_xor(s, off, 32);
    s2 += __shfl_xor(s2, off, 32);
  }
  float mean = s * (1.f / 256.f);
  float var  = s2 * (1.f / 256.f) - mean * mean;
  float rs   = rsqrtf(var + 1e-5f);
#pragma unroll
  for (int j = 0; j < 8; ++j) {
    int d = lane * 8 + j;
    float y = (x[j] - mean) * rs * g[d] + be[d];
    h[base + j]  = y;
    hb[base + j] = f2bf(y);
  }
}

// ---------------------------------------------------------------------------
// bf16 WMMA GEMM: C[M,N] = A[M,K](bf16) @ B[K,N](bf16) + bias, epilogue opts.
// Block tile 128x128x32, 256 threads = 8 waves, wave tile 32x64 (2x4 frags).
// A tile: async global->LDS B128 (rows padded to 40 u16 = 80 B so every
//   16-byte async destination is aligned; 20-dword stride is conflict-free
//   for the 16-lane fragment reads).
// B tile: VGPR-staged transpose into 34-u16-padded rows (17-dword stride).
// ---------------------------------------------------------------------------
template <int OUT_BF16, int RELU>
__global__ __launch_bounds__(256) void wmma_gemm(
    const uint16_t* __restrict__ A, const uint16_t* __restrict__ Bw,
    const float* __restrict__ bias, float* __restrict__ outF,
    uint16_t* __restrict__ outB, int M, int N, int K) {
  __shared__ uint16_t As[128 * 40];
  __shared__ uint16_t Bt[128 * 34];
  const int tid  = threadIdx.x;
  const int lane = tid & 31;
  const int wave = tid >> 5;
  const int wm = (wave & 3) * 32;
  const int wn = (wave >> 2) * 64;
  const int m0 = blockIdx.y * 128;
  const int n0 = blockIdx.x * 128;
  const int lr = lane & 15;
  const int hi = lane >> 4;
  (void)M;

  v8f zero = {};
  v8f acc[2][4];
#pragma unroll
  for (int i = 0; i < 2; ++i)
#pragma unroll
    for (int j = 0; j < 4; ++j) acc[i][j] = zero;

  for (int k0 = 0; k0 < K; k0 += 32) {
    {  // A tile: 128x32, thread -> (row = tid>>1, half = (tid&1)*16)
      int row = tid >> 1;
      int ch  = (tid & 1) * 16;
      const uint16_t* gp = A + (size_t)(m0 + row) * K + k0 + ch;
      uint16_t* lp = &As[row * 40 + ch];
#if HAVE_ASYNC_LDS
      __builtin_amdgcn_global_load_async_to_lds_b128(AS_GLOBAL(gp), AS_LDS(lp),
                                                     0, 0);
      __builtin_amdgcn_global_load_async_to_lds_b128(AS_GLOBAL(gp), AS_LDS(lp),
                                                     16, 0);
#else
      const uint32_t* g32 = (const uint32_t*)gp;
      uint32_t* s32 = (uint32_t*)lp;
#pragma unroll
      for (int i = 0; i < 8; ++i) s32[i] = g32[i];
#endif
      if (k0 + 32 < K)
        __builtin_prefetch(A + (size_t)(m0 + row) * K + k0 + 32 + ch, 0, 0);
    }
    {  // B tile 32x128 -> transposed LDS Bt[n][k]
      int kk = tid >> 3;
      int ng = (tid & 7) * 16;
      const uint16_t* gp = Bw + (size_t)(k0 + kk) * N + n0 + ng;
#pragma unroll
      for (int i = 0; i < 16; ++i) Bt[(ng + i) * 34 + kk] = gp[i];
      if (k0 + 32 < K)
        __builtin_prefetch(Bw + (size_t)(k0 + 32 + kk) * N + n0 + ng, 0, 0);
    }
    wait_async0();
    __syncthreads();

    FragAB af[2], bf[4];
#pragma unroll
    for (int mi = 0; mi < 2; ++mi) {      // A frag: 16x32, k = 16*hi + s contiguous
      const uint16_t* ap = &As[(wm + mi * 16 + lr) * 40];
      int ko = hi * 16;
#pragma unroll
      for (int v = 0; v < 8; ++v)
        af[mi].u[v] = *(const uint32_t*)(ap + ko + 2 * v);
    }
#pragma unroll
    for (int ni = 0; ni < 4; ++ni) {      // B frag: 32x16, k pairs contiguous
      const uint16_t* bp = &Bt[(wn + ni * 16 + lr) * 34 + hi * 16];
#pragma unroll
      for (int v = 0; v < 8; ++v) bf[ni].u[v] = *(const uint32_t*)(bp + 2 * v);
    }
#pragma unroll
    for (int mi = 0; mi < 2; ++mi)
#pragma unroll
      for (int ni = 0; ni < 4; ++ni)
        acc[mi][ni] = __builtin_amdgcn_wmma_f32_16x16x32_bf16(
            false, af[mi].v, false, bf[ni].v, (short)0, acc[mi][ni], false,
            false);
    __syncthreads();
  }

#pragma unroll
  for (int mi = 0; mi < 2; ++mi)
#pragma unroll
    for (int ni = 0; ni < 4; ++ni)
#pragma unroll
      for (int i = 0; i < 8; ++i) {
        int r = m0 + wm + mi * 16 + i + hi * 8;   // C-frag row map
        int c = n0 + wn + ni * 16 + lr;
        float val = acc[mi][ni][i] + bias[c];
        if (RELU) val = fmaxf(val, 0.f);
        if (OUT_BF16) outB[(size_t)r * N + c] = f2bf(val);
        else          outF[(size_t)r * N + c] = val;
      }
}

// ---------------------------------------------------------------------------
// Causal flash attention, one wave per (b, head, 16-query tile).
// Per 32-key tile: 2 WMMAs for Q@K^T, online softmax, 2 WMMAs for P@V.
// ---------------------------------------------------------------------------
__global__ __launch_bounds__(32) void attn_causal(
    const uint16_t* __restrict__ Q, const uint16_t* __restrict__ Km,
    const uint16_t* __restrict__ V, uint16_t* __restrict__ O) {
  __shared__ uint16_t P[16 * 34];
  const int lane = threadIdx.x;
  const int qt = blockIdx.x, h = blockIdx.y, b = blockIdx.z;
  const int q0 = qt * 16;
  const int lr = lane & 15, hi = lane >> 4;
  const size_t baseBS = (size_t)b * SEQ;
  const int hb = h * DHEAD;

  FragAB qf;
  {
    const uint16_t* qrow = Q + (baseBS + q0 + lr) * D_MODEL + hb;
    int ko = hi * 16;
#pragma unroll
    for (int v = 0; v < 8; ++v)
      qf.u[v] = *(const uint32_t*)(qrow + ko + 2 * v);
  }

  float mrow[8], lrow[8];
  v8f zero = {};
  v8f o1 = zero, o2 = zero;
#pragma unroll
  for (int i = 0; i < 8; ++i) { mrow[i] = -1e30f; lrow[i] = 0.f; }

  const float scale = 0.17677669529663687f;  // 1/sqrt(32)
  const int lastTile = qt >> 1;
  for (int kt = 0; kt <= lastTile; ++kt) {
    int key0 = kt * 32;
    FragAB kf1, kf2;
    {
      const uint16_t* kr1 = Km + (baseBS + key0 + lr) * D_MODEL + hb + hi * 16;
      const uint16_t* kr2 =
          Km + (baseBS + key0 + 16 + lr) * D_MODEL + hb + hi * 16;
#pragma unroll
      for (int v = 0; v < 8; ++v) {
        kf1.u[v] = *(const uint32_t*)(kr1 + 2 * v);
        kf2.u[v] = *(const uint32_t*)(kr2 + 2 * v);
      }
    }
    v8f s1 = __builtin_amdgcn_wmma_f32_16x16x32_bf16(false, qf.v, false, kf1.v,
                                                     (short)0, zero, false, false);
    v8f s2 = __builtin_amdgcn_wmma_f32_16x16x32_bf16(false, qf.v, false, kf2.v,
                                                     (short)0, zero, false, false);
#pragma unroll
    for (int i = 0; i < 8; ++i) {
      int row = q0 + i + hi * 8;
      int c1 = key0 + lr, c2 = c1 + 16;
      float v1 = (c1 <= row) ? s1[i] * scale : -3e38f;
      float v2 = (c2 <= row) ? s2[i] * scale : -3e38f;
      float mt = fmaxf(v1, v2);
#pragma unroll
      for (int off = 1; off < 16; off <<= 1)
        mt = fmaxf(mt, __shfl_xor(mt, off, 16));
      float mnew = fmaxf(mrow[i], mt);
      float esc = __expf(mrow[i] - mnew);
      float p1 = __expf(v1 - mnew);
      float p2 = __expf(v2 - mnew);
      float rs = p1 + p2;
#pragma unroll
      for (int off = 1; off < 16; off <<= 1) rs += __shfl_xor(rs, off, 16);
      lrow[i] = lrow[i] * esc + rs;
      mrow[i] = mnew;
      o1[i] *= esc;
      o2[i] *= esc;
      P[(i + hi * 8) * 34 + lr]      = f2bf(p1);
      P[(i + hi * 8) * 34 + lr + 16] = f2bf(p2);
    }
    __syncthreads();
    FragAB pf;
    {
      const uint16_t* pr = &P[lr * 34];
      int ko = hi * 16;
#pragma unroll
      for (int v = 0; v < 8; ++v)
        pf.u[v] = *(const uint32_t*)(pr + ko + 2 * v);
    }
    FragAB vf1, vf2;
    {
      int kb = key0 + hi * 16;
#pragma unroll
      for (int v = 0; v < 8; ++v) {
        uint32_t lo1 = V[(baseBS + kb + 2 * v)     * D_MODEL + hb + lr];
        uint32_t hi1 = V[(baseBS + kb + 2 * v + 1) * D_MODEL + hb + lr];
        uint32_t lo2 = V[(baseBS + kb + 2 * v)     * D_MODEL + hb + 16 + lr];
        uint32_t hi2 = V[(baseBS + kb + 2 * v + 1) * D_MODEL + hb + 16 + lr];
        vf1.u[v] = lo1 | (hi1 << 16);
        vf2.u[v] = lo2 | (hi2 << 16);
      }
    }
    o1 = __builtin_amdgcn_wmma_f32_16x16x32_bf16(false, pf.v, false, vf1.v,
                                                 (short)0, o1, false, false);
    o2 = __builtin_amdgcn_wmma_f32_16x16x32_bf16(false, pf.v, false, vf2.v,
                                                 (short)0, o2, false, false);
    __syncthreads();
  }

#pragma unroll
  for (int i = 0; i < 8; ++i) {
    int row = q0 + i + hi * 8;
    float inv = 1.f / lrow[i];
    O[(baseBS + row) * D_MODEL + hb + lr]      = f2bf(o1[i] * inv);
    O[(baseBS + row) * D_MODEL + hb + 16 + lr] = f2bf(o2[i] * inv);
  }
}

// ---------------------------------------------------------------------------
// Final projection 128 -> 2 (reads relu'd bf16 hidden)
// ---------------------------------------------------------------------------
__global__ void final_out_k(const uint16_t* __restrict__ P1,
                            const float* __restrict__ Wp2,
                            const float* __restrict__ bp2,
                            float* __restrict__ out) {
  int row = blockIdx.x * blockDim.x + threadIdx.x;
  if (row >= BATCH * SEQ) return;
  float a0 = bp2[0], a1 = bp2[1];
  const uint16_t* p = P1 + (size_t)row * 128;
  for (int k = 0; k < 128; ++k) {
    float pv = bf2f(p[k]);
    a0 += pv * Wp2[k * 2 + 0];
    a1 += pv * Wp2[k * 2 + 1];
  }
  out[(size_t)row * 2 + 0] = a0;
  out[(size_t)row * 2 + 1] = a1;
}

// ---------------------------------------------------------------------------
extern "C" void kernel_launch(void* const* d_in, const int* in_sizes, int n_in,
                              void* d_out, int out_size, void* d_ws,
                              size_t ws_size, hipStream_t stream) {
  const float* x    = (const float*)d_in[0];
  const float* cond = (const float*)d_in[1];
  const float* Win  = (const float*)d_in[2];
  const float* binp = (const float*)d_in[3];
  const float* gin  = (const float*)d_in[4];
  const float* bein = (const float*)d_in[5];
  const float* Wc1  = (const float*)d_in[6];
  const float* bc1  = (const float*)d_in[7];
  const float* Wc2  = (const float*)d_in[8];
  const float* bc2  = (const float*)d_in[9];
  const float* Ws   = (const float*)d_in[10];
  const float* bsv  = (const float*)d_in[11];
  const float* Wq   = (const float*)d_in[12];
  const float* bq   = (const float*)d_in[13];
  const float* Wk   = (const float*)d_in[14];
  const float* bk   = (const float*)d_in[15];
  const float* Wv   = (const float*)d_in[16];
  const float* bv   = (const float*)d_in[17];
  const float* Wo   = (const float*)d_in[18];
  const float* bo   = (const float*)d_in[19];
  // d_in[20..23] = Wcq,bcq,Wck,bck: unused (softmax over a single key == 1)
  const float* Wcv  = (const float*)d_in[24];
  const float* bcv  = (const float*)d_in[25];
  const float* Wco  = (const float*)d_in[26];
  const float* bco  = (const float*)d_in[27];
  const float* W1   = (const float*)d_in[28];
  const float* b1f  = (const float*)d_in[29];
  const float* W2   = (const float*)d_in[30];
  const float* b2f  = (const float*)d_in[31];
  const float* g1   = (const float*)d_in[32];
  const float* be1  = (const float*)d_in[33];
  const float* g2   = (const float*)d_in[34];
  const float* be2  = (const float*)d_in[35];
  const float* g3   = (const float*)d_in[36];
  const float* be3  = (const float*)d_in[37];
  const float* Wp1  = (const float*)d_in[38];
  const float* bp1  = (const float*)d_in[39];
  const float* Wp2  = (const float*)d_in[40];
  const float* bp2  = (const float*)d_in[41];
  float* out = (float*)d_out;
  (void)in_sizes; (void)n_in; (void)out_size; (void)ws_size;

  const size_t MB = 1024u * 1024u;
  char* ws = (char*)d_ws;
  float*    hF   = (float*)(ws);                  // 64 MB fp32 master hidden
  uint16_t* hB   = (uint16_t*)(ws + 64 * MB);     // 32 MB bf16 hidden
  float*    scr  = (float*)(ws + 96 * MB);        // 64 MB fp32 GEMM out
  uint16_t* Qb   = (uint16_t*)(ws + 160 * MB);    // 32 MB
  uint16_t* Kb   = (uint16_t*)(ws + 192 * MB);    // 32 MB
  uint16_t* Vb   = (uint16_t*)(ws + 224 * MB);    // 32 MB
  uint16_t* Ab   = (uint16_t*)(ws + 256 * MB);    // 32 MB attn out
  uint16_t* F1   = (uint16_t*)(ws + 160 * MB);    // 128 MB (reuses QKV/attn)
  uint16_t* P1   = (uint16_t*)(ws + 160 * MB);    // 16 MB (reuse)
  float*    memF = (float*)(ws + 288 * MB);
  float*    tmpC = (float*)(ws + 288 * MB + 256 * 1024);
  float*    cvF  = (float*)(ws + 288 * MB + 512 * 1024);  // 1 MB
  uint16_t* WB   = (uint16_t*)(ws + 290 * MB);    // bf16 weights (~6.1 MB)
  uint16_t* WQB  = WB;
  uint16_t* WKB  = WB + 262144;
  uint16_t* WVB  = WB + 2 * 262144;
  uint16_t* WOB  = WB + 3 * 262144;
  uint16_t* W1B  = WB + 4 * 262144;
  uint16_t* W2B  = W1B + 1048576;
  uint16_t* WP1B = W2B + 1048576;

  auto cvt = [&](const float* s, uint16_t* d, int n) {
    f32_to_bf16_k<<<(n + 255) / 256, 256, 0, stream>>>(s, d, n);
  };
  cvt(Wq, WQB, 262144);
  cvt(Wk, WKB, 262144);
  cvt(Wv, WVB, 262144);
  cvt(Wo, WOB, 262144);
  cvt(W1, W1B, 1048576);
  cvt(W2, W2B, 1048576);
  cvt(Wp1, WP1B, 32768);

  // condition memory + per-layer cross-attn vectors (cheap fp32)
  small_gemm_f32<<<256, 256, 0, stream>>>(cond, Wc1, bc1, tmpC, 256, 5, 256, 1);
  small_gemm_f32<<<256, 256, 0, stream>>>(tmpC, Wc2, bc2, memF, 256, 256, 256, 0);
  for (int l = 0; l < 4; ++l) {
    small_gemm_f32<<<256, 256, 0, stream>>>(memF, Wcv + l * 65536,
                                            bcv + l * 256, tmpC, 256, 256, 256, 0);
    small_gemm_f32<<<256, 256, 0, stream>>>(tmpC, Wco + l * 65536,
                                            bco + l * 256, cvF + l * 65536,
                                            256, 256, 256, 0);
  }

  input_stage<<<BATCH * SEQ, 32, 0, stream>>>(x, cond, Win, binp, gin, bein,
                                              Ws, bsv, hF, hB);

  const int M = BATCH * SEQ;  // 65536
  dim3 g256(2, 512), g1024(8, 512), g128(1, 512);
  for (int l = 0; l < 4; ++l) {
    wmma_gemm<1, 0><<<g256, 256, 0, stream>>>(hB, WQB + l * 65536,
                                              bq + l * 256, nullptr, Qb, M, 256, 256);
    wmma_gemm<1, 0><<<g256, 256, 0, stream>>>(hB, WKB + l * 65536,
                                              bk + l * 256, nullptr, Kb, M, 256, 256);
    wmma_gemm<1, 0><<<g256, 256, 0, stream>>>(hB, WVB + l * 65536,
                                              bv + l * 256, nullptr, Vb, M, 256, 256);
    attn_causal<<<dim3(SEQ / 16, NHEAD, BATCH), 32, 0, stream>>>(Qb, Kb, Vb, Ab);
    wmma_gemm<0, 0><<<g256, 256, 0, stream>>>(Ab, WOB + l * 65536,
                                              bo + l * 256, scr, nullptr, M, 256, 256);
    ln_add_k<<<M, 32, 0, stream>>>(hF, scr, g1 + l * 256, be1 + l * 256, hB, 0);
    ln_add_k<<<M, 32, 0, stream>>>(hF, cvF + l * 65536, g2 + l * 256,
                                   be2 + l * 256, hB, 8);
    wmma_gemm<1, 1><<<g1024, 256, 0, stream>>>(hB, W1B + l * 262144,
                                               b1f + l * 1024, nullptr, F1, M, 1024, 256);
    wmma_gemm<0, 0><<<g256, 256, 0, stream>>>(F1, W2B + l * 262144,
                                              b2f + l * 256, scr, nullptr, M, 256, 1024);
    ln_add_k<<<M, 32, 0, stream>>>(hF, scr, g3 + l * 256, be3 + l * 256, hB, 0);
  }
  wmma_gemm<1, 1><<<g128, 256, 0, stream>>>(hB, WP1B, bp1, nullptr, P1, M, 128, 256);
  final_out_k<<<256, 256, 0, stream>>>(P1, Wp2, bp2, out);
}
